// FusedConvInstanceNormReLU_14980845928807
// MI455X (gfx1250) — compile-verified, matrix operand order checked by reading, not run on hardware
//
#include <hip/hip_runtime.h>
#include <hip/hip_bf16.h>
#include <stdint.h>

// ---------------------------------------------------------------------------
// Fused Conv3x3(128->128, pad1) + bias + InstanceNorm + affine + ReLU
// Implicit GEMM on v_wmma_f32_16x16x32_bf16, async global->LDS DMA, wave32.
// ---------------------------------------------------------------------------

typedef __attribute__((ext_vector_type(16))) __bf16 v16bf;
typedef __attribute__((ext_vector_type(8)))  float  v8f;

#define A_STRIDE 80                 // bytes per A row (64B data + 16B pad: conflict-free, 16B aligned)
#define B_STRIDE 80
#define A_ROWS   130                // w' = -1 .. 128 (rows 0 and 129 are zero W-padding)
#define A_BUF    (A_ROWS * A_STRIDE)   // 10400 B
#define B_BUF    (128 * B_STRIDE)      // 10240 B
#define SMEM_BYTES (2 * A_BUF + 2 * B_BUF)  // 41280 B

#define XB_BYTES  (8u * 128u * 128u * 128u * 2u)   // bf16 NHWC x: 33,554,432
#define WP_BYTES  (9u * 128u * 128u * 2u)          // packed bf16 weights: 294,912

struct Frag { uint4 lo, hi; };
union FragCast { Frag f; v16bf v; };

__device__ __forceinline__ uint32_t lds_base_off(const void* p) {
  // generic -> LDS address space; ptrtoint of an AS(3) pointer is the LDS byte offset
  return (uint32_t)(uintptr_t)(__attribute__((address_space(3))) const char*)p;
}

__device__ __forceinline__ void async_copy_b128(uint32_t lds_addr, const void* gaddr) {
  asm volatile("global_load_async_to_lds_b128 %0, %1, off"
               :: "v"(lds_addr), "v"((uint64_t)(uintptr_t)gaddr)
               : "memory");
}

// ---------------------------------------------------------------------------
// prep: x NCHW f32 -> NHWC bf16 (LDS-tiled transpose over (c,w))
// ---------------------------------------------------------------------------
__global__ __launch_bounds__(256) void prep_x_kernel(const float* __restrict__ x,
                                                     __bf16* __restrict__ xb) {
  __shared__ float t[128][33];
  const int tid = threadIdx.x;
  const int cb = blockIdx.x & 3;
  const int h  = (blockIdx.x >> 2) & 127;
  const int n  = blockIdx.x >> 9;
#pragma unroll
  for (int i = 0; i < 16; ++i) {
    int e = tid + i * 256;              // 32c x 128w tile
    int cl = e >> 7, w = e & 127;
    t[w][cl] = x[(((size_t)(n * 128 + cb * 32 + cl) * 128) + h) * 128 + w];
  }
  __syncthreads();
#pragma unroll
  for (int i = 0; i < 16; ++i) {
    int e = tid + i * 256;
    int cl = e & 31, w = e >> 5;
    xb[(size_t)((n * 128 + h) * 128 + w) * 128 + cb * 32 + cl] = (__bf16)t[w][cl];
  }
}

// ---------------------------------------------------------------------------
// prep: weight OIHW f32 -> [kh][kw][cinblk][cout][cin32] bf16
// ---------------------------------------------------------------------------
__global__ __launch_bounds__(256) void prep_w_kernel(const float* __restrict__ w,
                                                     __bf16* __restrict__ wp) {
  int o = blockIdx.x * blockDim.x + threadIdx.x;
  if (o >= 9 * 128 * 128) return;
  int cil = o & 31;
  int co  = (o >> 5) & 127;
  int cb  = (o >> 12) & 3;
  int tap = o >> 14;
  int kh = tap / 3, kw = tap % 3;
  int ci = cb * 32 + cil;
  wp[o] = (__bf16)w[(((size_t)co * 128 + ci) * 3 + kh) * 3 + kw];
}

__global__ void zero_stats_kernel(float* __restrict__ s) {
  s[blockIdx.x * blockDim.x + threadIdx.x] = 0.0f;
}

// ---------------------------------------------------------------------------
// main: one WG per (n, h) row; 8 waves, each 64x32 of the 128x128 tile.
// ---------------------------------------------------------------------------
__global__ __launch_bounds__(256) void conv_wmma_kernel(
    const __bf16* __restrict__ xb,    // NHWC bf16
    const __bf16* __restrict__ wp,    // packed bf16 weights
    const float*  __restrict__ bias,
    float*        __restrict__ out,   // NCHW f32 (pre-norm conv output)
    float*        __restrict__ stats) // [8][128][2] sum / sumsq
{
  extern __shared__ char smem[];
  const int tid  = threadIdx.x;
  const int lane = tid & 31;
  const int wid  = tid >> 5;
  const int n = blockIdx.x >> 7;
  const int h = blockIdx.x & 127;
  const uint32_t lds0 = lds_base_off(smem);

  // Zero the W-padding rows (0 and 129) of both A buffers; never rewritten.
  if (tid < 80) {
    int buf = tid & 1, top = (tid >> 1) & 1, d = tid >> 2;
    *(uint32_t*)(smem + buf * A_BUF + top * (A_ROWS - 1) * A_STRIDE + d * 4) = 0u;
  }
  __syncthreads();

  // Valid kh taps are contiguous: kh0..kh0+nkh-1 (skip out-of-bounds H pad rows).
  // Branchless -> no local array -> no scratch spill in the hot loop.
  const int kh0  = (h == 0) ? 1 : 0;
  const int nkh  = 3 - ((h == 0) ? 1 : 0) - ((h == 127) ? 1 : 0);
  const int niter = nkh * 12;   // nkh x 4 cin-blocks x 3 kw

  auto issue = [&](int t, int buf) {
    const int kh = kh0 + t / 12, cb = (t / 3) & 3, kw = t % 3;
    const int hp = h + kh - 1;
    // A tile: 128 rows (w'=0..127) x 32 cin bf16, contiguous 64B per row in NHWC
    const __bf16* xrow = xb + (size_t)((n * 128 + hp) * 128) * 128;
    const uint32_t abase = lds0 + buf * A_BUF;
#pragma unroll
    for (int i = 0; i < 2; ++i) {
      int idx = tid + i * 256;            // 512 x 16B chunks
      int row = idx >> 2, part = idx & 3;
      async_copy_b128(abase + (row + 1) * A_STRIDE + part * 16,
                      xrow + row * 128 + cb * 32 + part * 8);
    }
    // B tile: 128 cout x 32 cin bf16, contiguous 8KB in packed layout
    const __bf16* wt = wp + (size_t)(((kh * 3 + kw) * 4 + cb) * 128) * 32;
    const uint32_t bbase = lds0 + 2 * A_BUF + buf * B_BUF;
#pragma unroll
    for (int i = 0; i < 2; ++i) {
      int idx = tid + i * 256;
      int col = idx >> 2, part = idx & 3;
      async_copy_b128(bbase + col * B_STRIDE + part * 16,
                      wt + col * 32 + part * 8);
    }
  };

  const int mrow = (wid & 1) * 64;        // wave's 64 output rows (pixels)
  const int ncol = (wid >> 1) * 32;       // wave's 32 output cols (cout)
  const int l15  = lane & 15;
  const int kh16 = (lane >> 4) * 16;      // byte offset selecting K-half per lane group

  v8f acc[8] = {};                        // 4 M-frags x 2 N-frags, f32 accum

  auto compute = [&](int t, int buf) {
    const int kw = t % 3;
    const char* ab = smem + buf * A_BUF;
    const char* bb = smem + 2 * A_BUF + buf * B_BUF;
    FragCast afr[4], bfr[2];
#pragma unroll
    for (int f = 0; f < 4; ++f) {
      // pixel m reads w' = m + kw - 1  ->  LDS row = m + kw
      uint32_t off = (uint32_t)((mrow + f * 16 + l15 + kw) * A_STRIDE + kh16);
      afr[f].f.lo = *(const uint4*)(ab + off);
      afr[f].f.hi = *(const uint4*)(ab + off + 32);
    }
#pragma unroll
    for (int g = 0; g < 2; ++g) {
      uint32_t off = (uint32_t)((ncol + g * 16 + l15) * B_STRIDE + kh16);
      bfr[g].f.lo = *(const uint4*)(bb + off);
      bfr[g].f.hi = *(const uint4*)(bb + off + 32);
    }
#pragma unroll
    for (int f = 0; f < 4; ++f)
#pragma unroll
      for (int g = 0; g < 2; ++g)
        acc[f * 2 + g] = __builtin_amdgcn_wmma_f32_16x16x32_bf16(
            false, afr[f].v, false, bfr[g].v, (short)0, acc[f * 2 + g],
            false, false);
  };

  // Software pipeline: double-buffered async DMA overlapped with WMMA.
  issue(0, 0);
  for (int t = 0; t < niter; ++t) {
    if (t + 1 < niter) {
      issue(t + 1, (t + 1) & 1);
      asm volatile("s_wait_asynccnt 0x4" ::: "memory"); // iter-t DMA done, t+1 in flight
    } else {
      asm volatile("s_wait_asynccnt 0x0" ::: "memory");
    }
    __syncthreads();
    compute(t, t & 1);
    __syncthreads();   // all reads of this buffer done before it is re-filled
  }

  // Epilogue: bias, store conv output, per-instance sum / sumsq.
#pragma unroll
  for (int g = 0; g < 2; ++g) {
    const int col = ncol + g * 16 + l15;
    const float bv = bias[col];
    float s1 = 0.0f, s2 = 0.0f;
#pragma unroll
    for (int f = 0; f < 4; ++f) {
      v8f a = acc[f * 2 + g];
      const int mbase = mrow + f * 16 + (lane >> 4) * 8;  // 8 consecutive w per lane
      float* op = out + (((size_t)(n * 128 + col) * 128 + h) * 128 + mbase);
      float y[8];
#pragma unroll
      for (int r = 0; r < 8; ++r) {
        y[r] = a[r] + bv;
        s1 += y[r];
        s2 += y[r] * y[r];
      }
      *(float4*)(op)     = make_float4(y[0], y[1], y[2], y[3]);
      *(float4*)(op + 4) = make_float4(y[4], y[5], y[6], y[7]);
    }
    s1 += __shfl_xor(s1, 16, 32);   // combine the two lane halves (same col)
    s2 += __shfl_xor(s2, 16, 32);
    if (lane < 16) {
      atomicAdd(&stats[(n * 128 + col) * 2 + 0], s1);
      atomicAdd(&stats[(n * 128 + col) * 2 + 1], s2);
    }
  }
}

// ---------------------------------------------------------------------------
// finalize: in-place InstanceNorm + affine + ReLU over d_out
// ---------------------------------------------------------------------------
__global__ __launch_bounds__(256) void finalize_kernel(
    float* __restrict__ out, const float* __restrict__ stats,
    const float* __restrict__ gamma, const float* __restrict__ beta) {
  const size_t idx  = (size_t)blockIdx.x * blockDim.x + threadIdx.x;
  const size_t base = idx * 4;                 // float4 per thread, same (n,c) row
  const int c = (int)((base >> 14) & 127);
  const int n = (int)(base >> 21);
  const float s1 = stats[(n * 128 + c) * 2 + 0];
  const float s2 = stats[(n * 128 + c) * 2 + 1];
  const float mean = s1 * (1.0f / 16384.0f);
  const float var  = fmaxf(s2 * (1.0f / 16384.0f) - mean * mean, 0.0f);
  const float inv  = rsqrtf(var + 1e-5f);
  const float sc = gamma[c] * inv;
  const float sh = beta[c] - mean * sc;
  float4 v = *(float4*)(out + base);
  v.x = fmaxf(v.x * sc + sh, 0.0f);
  v.y = fmaxf(v.y * sc + sh, 0.0f);
  v.z = fmaxf(v.z * sc + sh, 0.0f);
  v.w = fmaxf(v.w * sc + sh, 0.0f);
  *(float4*)(out + base) = v;
}

// ---------------------------------------------------------------------------
extern "C" void kernel_launch(void* const* d_in, const int* in_sizes, int n_in,
                              void* d_out, int out_size, void* d_ws, size_t ws_size,
                              hipStream_t stream) {
  (void)in_sizes; (void)n_in; (void)out_size; (void)ws_size;
  const float* x     = (const float*)d_in[0];
  const float* w     = (const float*)d_in[1];
  const float* bias  = (const float*)d_in[2];
  const float* gamma = (const float*)d_in[3];
  const float* beta  = (const float*)d_in[4];
  float* out = (float*)d_out;

  char* ws = (char*)d_ws;
  __bf16* xb    = (__bf16*)ws;                         // NHWC bf16 activations
  __bf16* wp    = (__bf16*)(ws + XB_BYTES);            // packed bf16 weights
  float*  stats = (float*)(ws + XB_BYTES + WP_BYTES);  // [8][128][2]

  zero_stats_kernel<<<8, 256, 0, stream>>>(stats);
  prep_x_kernel<<<8 * 128 * 4, 256, 0, stream>>>(x, xb);
  prep_w_kernel<<<(9 * 128 * 128 + 255) / 256, 256, 0, stream>>>(w, wp);
  conv_wmma_kernel<<<8 * 128, 256, SMEM_BYTES, stream>>>(xb, wp, bias, out, stats);
  finalize_kernel<<<(16777216 / 4) / 256, 256, 0, stream>>>(out, stats, gamma, beta);
}
